// GroupAttentionLayer_5205500362754
// MI455X (gfx1250) — matrix operand order repaired and checked by reading, not compiled, not run on hardware
//
#include <hip/hip_runtime.h>
#include <hip/hip_bf16.h>

// ---------------------------------------------------------------------------
// GroupAttentionLayer on MI455X (gfx1250, wave32, WMMA).
// Attention (S = Q K^T/8, per-block softmax, O += P V) AND all three convs
// (3x3 CBL_Q, 1x1 Conv_K, 1x1 CBL_O) run on v_wmma_f32_16x16x32_bf16 with
// f32 accumulation; BN / softmax reductions are fp32 VALU.
// ---------------------------------------------------------------------------

#define BATCH 4
#define HH 64
#define WW 64
#define CC 128
#define SP (HH * WW)            // 4096 spatial per batch
#define NPIX (BATCH * SP)       // 16384 pixels total
#define NBLK 64                 // 8x8 blocks of 8x8 pixels
#define EPS 1e-3f
#define ALPHA 0.1f

typedef __attribute__((ext_vector_type(16))) __bf16 v16bf;
typedef __attribute__((ext_vector_type(8)))  float  v8f;

__device__ __forceinline__ unsigned short f2bf(float f) {
    unsigned int u = __builtin_bit_cast(unsigned int, f);
    unsigned int r = (u + 0x7FFFu + ((u >> 16) & 1u)) >> 16;   // RNE truncation
    return (unsigned short)r;
}
__device__ __forceinline__ __bf16 bfbits(unsigned short s) {
    return __builtin_bit_cast(__bf16, s);
}

// A-fragment 16x32 bf16 (contiguous k order, same as B): lane row m = lane&15,
// element e -> k = e + 16*(lane>>4).  `base` = row ptr at k=0.
__device__ __forceinline__ v16bf load_afrag(const unsigned short* base, int hi) {
    v16bf a;
    #pragma unroll
    for (int e = 0; e < 16; ++e)
        a[e] = bfbits(base[e + hi * 16]);
    return a;
}

__device__ __forceinline__ v8f wmma_bf16(v16bf a, v16bf b, v8f c) {
    return __builtin_amdgcn_wmma_f32_16x16x32_bf16(false, a, false, b,
                                                   (short)0, c, false, false);
}

// ---------------------------------------------------------------------------
// Weight pack: W [tap][ci][co] f32 -> wt [tap][co][ci] bf16 (B-frag friendly).
// grid = taps*128 (tap,co), thread = ci.
// ---------------------------------------------------------------------------
__global__ __launch_bounds__(128) void wpack_kernel(
    const float* __restrict__ w, unsigned short* __restrict__ wt) {
    int tco = blockIdx.x;
    int tap = tco >> 7, co = tco & 127, ci = threadIdx.x;
    wt[(size_t)tco * CC + ci] = f2bf(w[(size_t)(tap * CC + ci) * CC + co]);
}

// ---------------------------------------------------------------------------
// Pack x: xbf[pix][c] (queries / conv input) and vt[b][blk][c][t] (values).
// ---------------------------------------------------------------------------
__global__ __launch_bounds__(128) void packx_kernel(
    const float* __restrict__ x, unsigned short* __restrict__ xbf,
    unsigned short* __restrict__ vt) {
    int pix = blockIdx.x, c = threadIdx.x;
    size_t idx = (size_t)pix * CC + c;
    unsigned short v = f2bf(x[idx]);
    xbf[idx] = v;
    int b = pix >> 12, sp = pix & (SP - 1);
    int i = sp >> 6, j = sp & 63;
    int blk = (i >> 3) * 8 + (j >> 3);
    int t = (i & 7) * 8 + (j & 7);
    vt[(((size_t)(b * NBLK + blk)) * CC + c) * 64 + t] = v;
}

// ---------------------------------------------------------------------------
// Pack keys: BN(q_pre) + LeakyReLU -> bf16.
// ---------------------------------------------------------------------------
__global__ __launch_bounds__(128) void packq_kernel(
    const float* __restrict__ qpre, const float* __restrict__ g,
    const float* __restrict__ bt, const float* __restrict__ mean,
    const float* __restrict__ invstd, unsigned short* __restrict__ kbf) {
    int pix = blockIdx.x, c = threadIdx.x;
    size_t idx = (size_t)pix * CC + c;
    float qv = g[c] * (qpre[idx] - mean[c]) * invstd[c] + bt[c];
    qv = qv > 0.f ? qv : ALPHA * qv;
    kbf[idx] = f2bf(qv);
}

// ---------------------------------------------------------------------------
// WMMA 1x1 conv: out[pix][co] = in[pix][:] . wt[co][:] + bias[co].
// 4 waves/block, wave = one 16-pixel tile; 4 k-chunks x 8 co-tiles = 32 wmma.
// ---------------------------------------------------------------------------
#define CV_WAVES 4
__global__ __launch_bounds__(CV_WAVES * 32) void conv1x1_wmma_kernel(
    const unsigned short* __restrict__ in, const unsigned short* __restrict__ wt,
    const float* __restrict__ bias, float* __restrict__ out) {
    const int lane = threadIdx.x & 31;
    const int tile = blockIdx.x * CV_WAVES + (threadIdx.x >> 5);
    const int p0 = tile * 16;
    const int col = lane & 15, hi = lane >> 4;

    v16bf A[4];
    const unsigned short* ip = in + (size_t)(p0 + col) * CC;
    #pragma unroll
    for (int kc = 0; kc < 4; ++kc) A[kc] = load_afrag(ip + kc * 32, hi);

    v8f O[8];
    #pragma unroll
    for (int i = 0; i < 8; ++i)
        #pragma unroll
        for (int r = 0; r < 8; ++r) O[i][r] = 0.f;

    #pragma unroll
    for (int ct = 0; ct < 8; ++ct) {
        const unsigned short* wp = wt + (size_t)(ct * 16 + col) * CC + hi * 16;
        #pragma unroll
        for (int kc = 0; kc < 4; ++kc)
            O[ct] = wmma_bf16(A[kc], *(const v16bf*)(wp + kc * 32), O[ct]);
    }
    #pragma unroll
    for (int ct = 0; ct < 8; ++ct)
        #pragma unroll
        for (int r = 0; r < 8; ++r)
            out[(size_t)(p0 + r + 8 * hi) * CC + ct * 16 + col] =
                O[ct][r] + bias[ct * 16 + col];
}

// ---------------------------------------------------------------------------
// WMMA 3x3 SAME conv (implicit GEMM, K = 9 taps x 128 ci).  A 16-pixel tile
// lies in one image row (W=64), so the dy bounds test is wave-uniform and dx
// edges are handled by per-lane zero data (EXEC stays all-1s for WMMA).
// ---------------------------------------------------------------------------
__global__ __launch_bounds__(CV_WAVES * 32) void conv3x3_wmma_kernel(
    const unsigned short* __restrict__ xbf, const unsigned short* __restrict__ wt,
    const float* __restrict__ bias, float* __restrict__ out) {
    const int lane = threadIdx.x & 31;
    const int tile = blockIdx.x * CV_WAVES + (threadIdx.x >> 5);
    const int p0 = tile * 16;
    const int b = p0 >> 12, sp = p0 & (SP - 1);
    const int i = sp >> 6, j0 = sp & 63;
    const int col = lane & 15, hi = lane >> 4;
    const int j = j0 + col;                       // this lane's pixel column

    v8f O[8];
    #pragma unroll
    for (int t = 0; t < 8; ++t)
        #pragma unroll
        for (int r = 0; r < 8; ++r) O[t][r] = 0.f;

    #pragma unroll
    for (int dy = 0; dy < 3; ++dy) {
        int yy = i + dy - 1;
        if (yy < 0 || yy >= HH) continue;         // uniform across the wave
        #pragma unroll
        for (int dx = 0; dx < 3; ++dx) {
            int xx = j + dx - 1;
            bool valid = (xx >= 0) && (xx < WW);
            const unsigned short* ip =
                xbf + (size_t)(b * SP + yy * WW + (valid ? xx : 0)) * CC;
            v16bf A[4];
            #pragma unroll
            for (int kc = 0; kc < 4; ++kc) {
                v16bf a = load_afrag(ip + kc * 32, hi);
                if (!valid) {
                    #pragma unroll
                    for (int e = 0; e < 16; ++e) a[e] = bfbits(0);
                }
                A[kc] = a;
            }
            const unsigned short* wp0 = wt + (size_t)(dy * 3 + dx) * CC * CC;
            #pragma unroll
            for (int ct = 0; ct < 8; ++ct) {
                const unsigned short* wp =
                    wp0 + (size_t)(ct * 16 + col) * CC + hi * 16;
                #pragma unroll
                for (int kc = 0; kc < 4; ++kc)
                    O[ct] = wmma_bf16(A[kc], *(const v16bf*)(wp + kc * 32),
                                      O[ct]);
            }
        }
    }
    #pragma unroll
    for (int ct = 0; ct < 8; ++ct)
        #pragma unroll
        for (int r = 0; r < 8; ++r)
            out[(size_t)(p0 + r + 8 * hi) * CC + ct * 16 + col] =
                O[ct][r] + bias[ct * 16 + col];
}

// ---------------------------------------------------------------------------
// Per-channel mean / 1/sqrt(var+eps) over (B,H,W). One block per channel.
// ---------------------------------------------------------------------------
__global__ __launch_bounds__(256) void stats_kernel(
    const float* __restrict__ in, float* __restrict__ mean,
    float* __restrict__ invstd) {
    int c = blockIdx.x, t = threadIdx.x;
    float s = 0.f, ss = 0.f;
    for (int e = t; e < NPIX; e += 256) {
        float v = in[(size_t)e * CC + c];
        s += v; ss += v * v;
    }
    __shared__ float sh[256], sh2[256];
    sh[t] = s; sh2[t] = ss; __syncthreads();
    for (int off = 128; off > 0; off >>= 1) {
        if (t < off) { sh[t] += sh[t + off]; sh2[t] += sh2[t + off]; }
        __syncthreads();
    }
    if (t == 0) {
        float m = sh[0] * (1.f / NPIX);
        float var = sh2[0] * (1.f / NPIX) - m * m;
        mean[c] = m;
        invstd[c] = rsqrtf(var + EPS);
    }
}

// ---------------------------------------------------------------------------
// WMMA block attention. 4 waves/block, each wave owns a 16-query tile and
// walks all 64 KV blocks. Epilogue: y1 = acc + agg (in place on acc buffer).
//
// Fragment maps used here (wave32):
//   A 16x32 bf16 : m = lane&15,           k = e + 16*(lane>>4)
//   B 32x16 bf16 : n = lane&15,           k = e + 16*(lane>>4)
//   C/D 16x16 f32: m = vgpr + 8*(lane>>4), n = lane&15
// ---------------------------------------------------------------------------
#define AT_WAVES 4
#define PSTRIDE 72

__global__ __launch_bounds__(AT_WAVES * 32) void attn_kernel(
    const unsigned short* __restrict__ xbf,  // queries  [B*SP][C]
    const unsigned short* __restrict__ kbf,  // keys     [B*SP][C]
    const unsigned short* __restrict__ vt,   // values   [B*NBLK][C][64]
    float* __restrict__ y1)                  // in: acc, out: acc + agg
{
    __shared__ unsigned short ldsP[AT_WAVES][16 * PSTRIDE];
    const int lane = threadIdx.x & 31;
    const int wv = threadIdx.x >> 5;
    const int tile = blockIdx.x * AT_WAVES + wv;
    const int b = tile >> 8;                 // 256 query tiles per batch
    const int q0 = (tile & 255) * 16;
    const int col = lane & 15;
    const int hi = lane >> 4;

    v16bf Qf[4];
    const unsigned short* qp = xbf + (size_t)(b * SP + q0 + col) * CC;
    #pragma unroll
    for (int kc = 0; kc < 4; ++kc) Qf[kc] = load_afrag(qp + kc * 32, hi);

    v8f O[8];
    #pragma unroll
    for (int i = 0; i < 8; ++i)
        #pragma unroll
        for (int r = 0; r < 8; ++r) O[i][r] = 0.f;

    unsigned short* lp = ldsP[wv];

    for (int blk = 0; blk < NBLK; ++blk) {
        const int bn = blk >> 3, bm = blk & 7;

        // ---- scores S = Q K^T (4 key tiles of 16, 4 k-chunks of 32) ----
        v8f S[4];
        #pragma unroll
        for (int nt = 0; nt < 4; ++nt) {
            int t = nt * 16 + col;                         // key for this col
            int pix = (bn * 8 + (t >> 3)) * WW + bm * 8 + (t & 7);
            const unsigned short* kp =
                kbf + (size_t)(b * SP + pix) * CC + hi * 16;
            v8f c;
            #pragma unroll
            for (int r = 0; r < 8; ++r) c[r] = 0.f;
            #pragma unroll
            for (int kc = 0; kc < 4; ++kc)
                c = wmma_bf16(Qf[kc], *(const v16bf*)(kp + kc * 32), c);
            S[nt] = c;
        }

        // ---- per-row softmax over 64 keys (scale 1/8 folded into exp) ----
        #pragma unroll
        for (int r = 0; r < 8; ++r) {
            float v = fmaxf(fmaxf(S[0][r], S[1][r]), fmaxf(S[2][r], S[3][r]));
            v = fmaxf(v, __shfl_xor(v, 1, 32));
            v = fmaxf(v, __shfl_xor(v, 2, 32));
            v = fmaxf(v, __shfl_xor(v, 4, 32));
            v = fmaxf(v, __shfl_xor(v, 8, 32));
            float s = 0.f;
            #pragma unroll
            for (int nt = 0; nt < 4; ++nt) {
                float p = __expf(0.125f * (S[nt][r] - v));
                S[nt][r] = p; s += p;
            }
            s += __shfl_xor(s, 1, 32);
            s += __shfl_xor(s, 2, 32);
            s += __shfl_xor(s, 4, 32);
            s += __shfl_xor(s, 8, 32);
            float inv = 1.f / s;
            #pragma unroll
            for (int nt = 0; nt < 4; ++nt) S[nt][r] *= inv;
        }

        // ---- re-layout P: D-fragment -> LDS -> A-fragment (bf16) ----
        #pragma unroll
        for (int nt = 0; nt < 4; ++nt)
            #pragma unroll
            for (int r = 0; r < 8; ++r)
                lp[(r + 8 * hi) * PSTRIDE + nt * 16 + col] = f2bf(S[nt][r]);
        // same-wave LDS ops complete in order; reads below see the stores
        v16bf Pa[2];
        #pragma unroll
        for (int kch = 0; kch < 2; ++kch) {
            v16bf a;
            #pragma unroll
            for (int e = 0; e < 16; ++e) {
                int k = kch * 32 + e + hi * 16;
                a[e] = bfbits(lp[col * PSTRIDE + k]);
            }
            Pa[kch] = a;
        }

        // ---- O += P V (8 channel tiles x 2 key chunks) ----
        const unsigned short* vp = vt + (size_t)(b * NBLK + blk) * CC * 64;
        #pragma unroll
        for (int ct = 0; ct < 8; ++ct) {
            const unsigned short* vpc =
                vp + (size_t)(ct * 16 + col) * 64 + hi * 16;
            #pragma unroll
            for (int kch = 0; kch < 2; ++kch)
                O[ct] = wmma_bf16(Pa[kch], *(const v16bf*)(vpc + kch * 32),
                                  O[ct]);
        }
    }

    // ---- epilogue: y1 = acc + agg (in place) ----
    #pragma unroll
    for (int ct = 0; ct < 8; ++ct)
        #pragma unroll
        for (int r = 0; r < 8; ++r) {
            size_t idx =
                (size_t)(b * SP + q0 + r + 8 * hi) * CC + ct * 16 + col;
            y1[idx] += O[ct][r];
        }
}

// ---------------------------------------------------------------------------
// Per-(b,c) max of BN(y1) over spatial.  grid = B*C, 256 threads.
// ---------------------------------------------------------------------------
__global__ __launch_bounds__(256) void bcmax_kernel(
    const float* __restrict__ y1, const float* __restrict__ g,
    const float* __restrict__ bt, const float* __restrict__ mean,
    const float* __restrict__ invstd, float* __restrict__ mx) {
    int bc = blockIdx.x, b = bc >> 7, c = bc & 127, t = threadIdx.x;
    float gm = g[c] * invstd[c];
    float bb = bt[c] - gm * mean[c];
    float v = -3.4e38f;
    for (int e = t; e < SP; e += 256)
        v = fmaxf(v, gm * y1[(size_t)(b * SP + e) * CC + c] + bb);
    __shared__ float sh[256];
    sh[t] = v; __syncthreads();
    for (int off = 128; off > 0; off >>= 1) {
        if (t < off) sh[t] = fmaxf(sh[t], sh[t + off]);
        __syncthreads();
    }
    if (t == 0) mx[bc] = sh[0];
}

// ---------------------------------------------------------------------------
// e = exp(BN(y1) - max) (stored f32), plus per-(b,c) sum.
// ---------------------------------------------------------------------------
__global__ __launch_bounds__(256) void expsum_kernel(
    const float* __restrict__ y1, const float* __restrict__ g,
    const float* __restrict__ bt, const float* __restrict__ mean,
    const float* __restrict__ invstd, const float* __restrict__ mx,
    float* __restrict__ eout, float* __restrict__ sum) {
    int bc = blockIdx.x, b = bc >> 7, c = bc & 127, t = threadIdx.x;
    float gm = g[c] * invstd[c];
    float bb = bt[c] - gm * mean[c];
    float m = mx[bc];
    float s = 0.f;
    for (int e = t; e < SP; e += 256) {
        size_t idx = (size_t)(b * SP + e) * CC + c;
        float p = __expf(gm * y1[idx] + bb - m);
        eout[idx] = p; s += p;
    }
    __shared__ float sh[256];
    sh[t] = s; __syncthreads();
    for (int off = 128; off > 0; off >>= 1) {
        if (t < off) sh[t] += sh[t + off];
        __syncthreads();
    }
    if (t == 0) sum[bc] = sh[0];
}

// y3 = e / sum, re-packed to bf16 for the WMMA output conv.
__global__ __launch_bounds__(128) void normbf_kernel(
    const float* __restrict__ e, const float* __restrict__ sum,
    unsigned short* __restrict__ ebf) {
    int pix = blockIdx.x, c = threadIdx.x;
    int b = pix >> 12;
    ebf[(size_t)pix * CC + c] =
        f2bf(e[(size_t)pix * CC + c] / sum[b * CC + c]);
}

__global__ __launch_bounds__(128) void bnl_kernel(
    const float* __restrict__ in, const float* __restrict__ g,
    const float* __restrict__ bt, const float* __restrict__ mean,
    const float* __restrict__ invstd, float* __restrict__ out) {
    int pix = blockIdx.x, c = threadIdx.x;
    size_t idx = (size_t)pix * CC + c;
    float y = g[c] * (in[idx] - mean[c]) * invstd[c] + bt[c];
    out[idx] = y > 0.f ? y : ALPHA * y;
}

// ---------------------------------------------------------------------------
extern "C" void kernel_launch(void* const* d_in, const int* in_sizes, int n_in,
                              void* d_out, int out_size, void* d_ws,
                              size_t ws_size, hipStream_t stream) {
    const float* x   = (const float*)d_in[0];
    const float* Wq  = (const float*)d_in[1];
    const float* bq  = (const float*)d_in[2];
    const float* gq  = (const float*)d_in[3];
    const float* btq = (const float*)d_in[4];
    const float* Wk  = (const float*)d_in[5];
    const float* bk  = (const float*)d_in[6];
    const float* g1  = (const float*)d_in[7];
    const float* bt1 = (const float*)d_in[8];
    const float* Wo  = (const float*)d_in[9];
    const float* bo  = (const float*)d_in[10];
    const float* go  = (const float*)d_in[11];
    const float* bto = (const float*)d_in[12];
    float* out = (float*)d_out;

    char* ws = (char*)d_ws;
    const size_t F32 = (size_t)NPIX * CC * 4;   // 8 MiB
    const size_t BF = (size_t)NPIX * CC * 2;    // 4 MiB
    float* qpre = (float*)(ws);                          // -> e -> opre
    float* acc  = (float*)(ws + F32);                    // -> y1
    unsigned short* kbf = (unsigned short*)(ws + 2 * F32);        // -> ebf
    unsigned short* xbf = (unsigned short*)(ws + 2 * F32 + BF);
    unsigned short* vt  = (unsigned short*)(ws + 2 * F32 + 2 * BF);
    char* wbase = ws + 2 * F32 + 3 * BF;                 // 28 MiB offset
    unsigned short* wqt = (unsigned short*)(wbase);               // 288 KiB
    unsigned short* wkt = (unsigned short*)(wbase + 9 * CC * CC * 2);
    unsigned short* wot = (unsigned short*)(wbase + 10 * CC * CC * 2);
    float* st = (float*)(wbase + 11 * CC * CC * 2);
    float* meanq = st, * invq = st + 128;
    float* mean1 = st + 256, * inv1 = st + 384;
    float* meano = st + 512, * invo = st + 640;
    float* mx = st + 768, * sm = st + 1280;

    const int NTILE = NPIX / 16;                // 1024 pixel tiles

    // 0. pack weights (transposed bf16) and inputs
    wpack_kernel<<<9 * CC, 128, 0, stream>>>(Wq, wqt);
    wpack_kernel<<<CC, 128, 0, stream>>>(Wk, wkt);
    wpack_kernel<<<CC, 128, 0, stream>>>(Wo, wot);
    packx_kernel<<<NPIX, 128, 0, stream>>>(x, xbf, vt);
    // 1. q_pre = conv3x3(x, Wq) + bq ; acc = conv1x1(x, Wk) + bk   (WMMA)
    conv3x3_wmma_kernel<<<NTILE / CV_WAVES, CV_WAVES * 32, 0, stream>>>(
        xbf, wqt, bq, qpre);
    conv1x1_wmma_kernel<<<NTILE / CV_WAVES, CV_WAVES * 32, 0, stream>>>(
        xbf, wkt, bk, acc);
    // 2. BN stats of q_pre, pack bf16 keys
    stats_kernel<<<CC, 256, 0, stream>>>(qpre, meanq, invq);
    packq_kernel<<<NPIX, 128, 0, stream>>>(qpre, gq, btq, meanq, invq, kbf);
    // 3. WMMA block attention; y1 = acc + agg (in place on acc)
    attn_kernel<<<NTILE / AT_WAVES, AT_WAVES * 32, 0, stream>>>(
        xbf, kbf, vt, acc);
    // 4. y = BN(y1); spatial softmax per (b, c); repack bf16
    stats_kernel<<<CC, 256, 0, stream>>>(acc, mean1, inv1);
    bcmax_kernel<<<BATCH * CC, 256, 0, stream>>>(acc, g1, bt1, mean1, inv1, mx);
    expsum_kernel<<<BATCH * CC, 256, 0, stream>>>(acc, g1, bt1, mean1, inv1,
                                                  mx, qpre, sm);
    normbf_kernel<<<NPIX, 128, 0, stream>>>(qpre, sm, kbf);
    // 5. CBL_O: WMMA conv1x1 + BN + LeakyReLU
    conv1x1_wmma_kernel<<<NTILE / CV_WAVES, CV_WAVES * 32, 0, stream>>>(
        kbf, wot, bo, qpre);
    stats_kernel<<<CC, 256, 0, stream>>>(qpre, meano, invo);
    bnl_kernel<<<NPIX, 128, 0, stream>>>(qpre, go, bto, meano, invo, out);
}
